// TSGCN_seg_80917183857422
// MI455X (gfx1250) — matrix operand order repaired and checked by reading, not compiled, not run on hardware
//
#include <hip/hip_runtime.h>
#include <math.h>
#include <stdint.h>

// ---------------------------------------------------------------------------
// TSGCN segmentation forward for MI455X (gfx1250, wave32, WMMA).
//
//  * Conv-1x1 layers: WMMA GEMMs. Activations f16 [m][n][ci] (ci contiguous,
//    padded to 32) so each lane fills its fragment with contiguous
//    global_load_b128s per the CDNA5 16-bit A(16x32)/B(32x16) layouts.
//    Each wave computes a 16co x 32n tile (one A fragment, two B fragments,
//    two v_wmma_f32_16x16x32_f16 per k-step).
//  * KNN: Gram matrix on the matrix cores (d2 = |xi|^2+|xj|^2-2 xi.xj, dot
//    products via f16 WMMA w/ f32 accumulate), then a selection-only top-32
//    kernel that stages each 16KB distance row into LDS with
//    global_load_async_to_lds_b128 + s_wait_asynccnt (gfx1250 async path).
//  * BN / lrelu / channel-softmax / K-reductions fused into epilogues: model
//    is HBM/L2 bound (~150 GFLOP vs ~3 GB traffic @ 23.3 TB/s).
// ---------------------------------------------------------------------------

typedef __attribute__((ext_vector_type(16))) _Float16 v16h;
typedef __attribute__((ext_vector_type(8)))  _Float16 v8h;
typedef __attribute__((ext_vector_type(8)))  float    v8f;

#define NPTS 4096
#define KNN  32
#define BSZ  2

__device__ __forceinline__ float bn_lrelu(float v, float m, float r, float g, float b) {
  float y = (v - m) * r * g + b;
  return y > 0.f ? y : 0.01f * y;
}

// ---------------------------------------------------------------------------
// Weight packing: f32 (CO,CI) -> f16 (COpad, CKpad), zero padded.
// ---------------------------------------------------------------------------
__global__ void pack_w16(const float* __restrict__ w, _Float16* __restrict__ o,
                         int CO, int CI, int COpad, int CKpad) {
  int t = blockIdx.x * blockDim.x + threadIdx.x;
  if (t >= COpad * CKpad) return;
  int co = t / CKpad, c = t % CKpad;
  o[t] = (co < CO && c < CI) ? (_Float16)w[co * CI + c] : (_Float16)0.0f;
}

__global__ void pack_bias(const float* __restrict__ b, float* __restrict__ o,
                          int CO, int COpad) {
  int t = blockIdx.x * blockDim.x + threadIdx.x;
  if (t >= COpad) return;
  o[t] = (t < CO) ? b[t] : 0.0f;
}

// ---------------------------------------------------------------------------
// KNN prep: strided f32 (B,ctot,N) channels -> f16 [b][n][CKpad] (padded) and
// per-point squared norms xx[b][n] in f32.
// ---------------------------------------------------------------------------
__global__ void knn_prep(const float* __restrict__ src, int ctot, int coff,
                         int C, int CKpad,
                         _Float16* __restrict__ q16, float* __restrict__ xx) {
  int t = blockIdx.x * blockDim.x + threadIdx.x;
  if (t >= BSZ * NPTS) return;
  int b = t >> 12, n = t & (NPTS - 1);
  const float* sb = src + ((size_t)b * ctot + coff) * NPTS + n;
  _Float16* o = q16 + (size_t)t * CKpad;
  float s = 0.f;
  for (int c = 0; c < C; ++c) {
    float v = sb[(size_t)c * NPTS];
    s += v * v;
    o[c] = (_Float16)v;
  }
  for (int c = C; c < CKpad; ++c) o[c] = (_Float16)0.0f;
  xx[t] = s;
}

// ---------------------------------------------------------------------------
// KNN Gram: dist[b][q][j] = xx[q] + xx[j] - 2 * dot(x_q, x_j), dot via WMMA.
// grid (N/16 query tiles, N/128 candidate tiles, B); block 128 = 4 waves,
// each wave produces a 16q x 32j tile.
// ---------------------------------------------------------------------------
__global__ void knn_gram(const _Float16* __restrict__ Q, const float* __restrict__ xx,
                         int CK, float* __restrict__ dist) {
  const int b    = blockIdx.z;
  const int q0   = blockIdx.x << 4;
  const int wave = threadIdx.x >> 5;
  const int lane = threadIdx.x & 31;
  const int l    = lane & 15;
  const int h    = lane >> 4;
  const int j0   = (blockIdx.y << 7) + (wave << 5);

  const _Float16* qrow = Q + ((size_t)b * NPTS + q0 + l) * CK;
  const _Float16* c0   = Q + ((size_t)b * NPTS + j0 + l) * CK;
  const _Float16* c1   = c0 + (size_t)16 * CK;

  v8f acc0 = {}, acc1 = {};
#pragma unroll 2
  for (int k0 = 0; k0 < CK; k0 += 32) {
    v8h alo = *(const v8h*)(qrow + k0 + 8 * h);
    v8h ahi = *(const v8h*)(qrow + k0 + 16 + 8 * h);
    v16h a = __builtin_shufflevector(alo, ahi, 0,1,2,3,4,5,6,7,8,9,10,11,12,13,14,15);
    v8h b0lo = *(const v8h*)(c0 + k0 + 8 * h);
    v8h b0hi = *(const v8h*)(c0 + k0 + 16 + 8 * h);
    v16h b0 = __builtin_shufflevector(b0lo, b0hi, 0,1,2,3,4,5,6,7,8,9,10,11,12,13,14,15);
    acc0 = __builtin_amdgcn_wmma_f32_16x16x32_f16(false, a, false, b0,
                                                  (short)0, acc0, false, false);
    v8h b1lo = *(const v8h*)(c1 + k0 + 8 * h);
    v8h b1hi = *(const v8h*)(c1 + k0 + 16 + 8 * h);
    v16h b1 = __builtin_shufflevector(b1lo, b1hi, 0,1,2,3,4,5,6,7,8,9,10,11,12,13,14,15);
    acc1 = __builtin_amdgcn_wmma_f32_16x16x32_f16(false, a, false, b1,
                                                  (short)0, acc1, false, false);
  }
  const float* xb = xx + (size_t)b * NPTS;
  float xj0 = xb[j0 + l];
  float xj1 = xb[j0 + 16 + l];
#pragma unroll
  for (int r = 0; r < 8; ++r) {
    int q = q0 + r + 8 * h;
    float xq = xb[q];
    size_t row = ((size_t)b * NPTS + q) * NPTS;
    dist[row + j0 + l]      = xq + xj0 - 2.0f * acc0[r];
    dist[row + j0 + 16 + l] = xq + xj1 - 2.0f * acc1[r];
  }
}

// ---------------------------------------------------------------------------
// KNN select: stage one 4096-float distance row into LDS via async-to-LDS
// (gfx1250), then 32 rounds of keyed argmin ((d2_bits<<32)|j -> ties pick
// the lowest index, matching jax.lax.top_k on pd = -d2).
// ---------------------------------------------------------------------------
__global__ void knn_select(const float* __restrict__ dist, int* __restrict__ idx) {
  __shared__ __align__(16) float drow[NPTS];
  __shared__ unsigned long long red[256];
  const int nq = blockIdx.x, b = blockIdx.y, tid = threadIdx.x;
  const float* g = dist + ((size_t)b * NPTS + nq) * NPTS;
#if defined(__gfx1250__)
  for (int j = tid * 4; j < NPTS; j += 256 * 4) {
    unsigned ldsOff = (unsigned)(uintptr_t)(&drow[j]);
    asm volatile("global_load_async_to_lds_b128 %0, %1, off"
                 :: "v"(ldsOff), "v"(g + j) : "memory");
  }
  asm volatile("s_wait_asynccnt 0" ::: "memory");
#else
  for (int j = tid; j < NPTS; j += 256) drow[j] = g[j];
#endif
  __syncthreads();
  for (int kk = 0; kk < KNN; ++kk) {
    unsigned long long best = ~0ull;
    for (int j = tid; j < NPTS; j += 256) {
      unsigned long long key =
          ((unsigned long long)__float_as_uint(drow[j]) << 32) | (unsigned)j;
      best = best < key ? best : key;
    }
    red[tid] = best;
    __syncthreads();
    for (int s = 128; s > 0; s >>= 1) {
      if (tid < s) red[tid] = red[tid] < red[tid + s] ? red[tid] : red[tid + s];
      __syncthreads();
    }
    if (tid == 0) {
      int sel = (int)(red[0] & 0xffffffffu);
      idx[((size_t)b * NPTS + nq) * KNN + kk] = sel;
      drow[sel] = __builtin_inff();
    }
    __syncthreads();
  }
}

// ---------------------------------------------------------------------------
// Graph-feature gather: builds the three GEMM inputs for one stage directly
// in WMMA-friendly f16 [m=b*K+k][n][ci] layout (CKpad channels, zero padded):
//   cap = [nbr_cap, ctr_cap]; attn = [nbr_cap, nbr_cap-ctr_cap];
//   nrm = [nbr_nrm, ctr_nrm]
// ---------------------------------------------------------------------------
__global__ void gather_feat(const float* __restrict__ capS, int capCtot, int capCoff,
                            const float* __restrict__ nrmS, int nrmCtot, int nrmCoff,
                            const int* __restrict__ idx, int C, int CKpad,
                            _Float16* __restrict__ gCap,
                            _Float16* __restrict__ gAtt,
                            _Float16* __restrict__ gNrm) {
  const int b = blockIdx.z, k = blockIdx.y, n0 = blockIdx.x * 16;
  const int m = b * KNN + k;
  const int tasks = 16 * C;
  for (int t = threadIdx.x; t < tasks; t += blockDim.x) {
    int c = t % C, nl = t / C;
    int n = n0 + nl;
    int j = idx[((size_t)b * NPTS + n) * KNN + k];
    const float* cb = capS + (size_t)(b * capCtot + capCoff + c) * NPTS;
    const float* nb = nrmS + (size_t)(b * nrmCtot + nrmCoff + c) * NPTS;
    float nc = cb[j], cc = cb[n];
    float nn = nb[j], cn = nb[n];
    size_t base = ((size_t)m * NPTS + n) * CKpad;
    gCap[base + c]     = (_Float16)nc;
    gCap[base + C + c] = (_Float16)cc;
    gAtt[base + c]     = (_Float16)nc;
    gAtt[base + C + c] = (_Float16)(nc - cc);
    gNrm[base + c]     = (_Float16)nn;
    gNrm[base + C + c] = (_Float16)cn;
  }
  const int pw = CKpad - 2 * C;  // zero padding (stage 1: 24 -> 32)
  for (int t = threadIdx.x; t < 16 * pw; t += blockDim.x) {
    int c = 2 * C + t % pw, nl = t / pw;
    size_t base = ((size_t)m * NPTS + (n0 + nl)) * CKpad;
    gCap[base + c] = (_Float16)0.0f;
    gAtt[base + c] = (_Float16)0.0f;
    gNrm[base + c] = (_Float16)0.0f;
  }
}

// ---------------------------------------------------------------------------
// WMMA GEMM: Y[m,co,n] = sum_ci W16[co,ci] * X16[m,n,ci] + bias[co]
// grid = (N/128, CO/16, M); block = 128 (4 waves, each a 16co x 32n tile:
// one A fragment reused by two B fragments / two accumulators).
// ---------------------------------------------------------------------------
__global__ void conv_gemm_wmma(const _Float16* __restrict__ W,
                               const float* __restrict__ bias,
                               const _Float16* __restrict__ X,
                               float* __restrict__ Y,
                               int CK, int CO) {
  const int m    = blockIdx.z;
  const int co0  = blockIdx.y << 4;
  const int wave = threadIdx.x >> 5;
  const int lane = threadIdx.x & 31;
  const int l    = lane & 15;
  const int h    = lane >> 4;
  const int n0   = (blockIdx.x << 7) + (wave << 5);

  const _Float16* wrow  = W + (size_t)(co0 + l) * CK;
  const _Float16* xcol0 = X + ((size_t)m * NPTS + (n0 + l)) * CK;
  const _Float16* xcol1 = xcol0 + (size_t)16 * CK;

  v8f acc0 = {}, acc1 = {};
#pragma unroll 2
  for (int k0 = 0; k0 < CK; k0 += 32) {
    if (k0 + 32 < CK) {
      __builtin_prefetch((const void*)(xcol0 + k0 + 32), 0, 1);
      __builtin_prefetch((const void*)(xcol1 + k0 + 32), 0, 1);
    }
    v8h alo = *(const v8h*)(wrow + k0 + 8 * h);
    v8h ahi = *(const v8h*)(wrow + k0 + 16 + 8 * h);
    v16h a = __builtin_shufflevector(alo, ahi, 0,1,2,3,4,5,6,7,8,9,10,11,12,13,14,15);
    v8h b0lo = *(const v8h*)(xcol0 + k0 + 8 * h);
    v8h b0hi = *(const v8h*)(xcol0 + k0 + 16 + 8 * h);
    v16h b0 = __builtin_shufflevector(b0lo, b0hi, 0,1,2,3,4,5,6,7,8,9,10,11,12,13,14,15);
    acc0 = __builtin_amdgcn_wmma_f32_16x16x32_f16(false, a, false, b0,
                                                  (short)0, acc0, false, false);
    v8h b1lo = *(const v8h*)(xcol1 + k0 + 8 * h);
    v8h b1hi = *(const v8h*)(xcol1 + k0 + 16 + 8 * h);
    v16h b1 = __builtin_shufflevector(b1lo, b1hi, 0,1,2,3,4,5,6,7,8,9,10,11,12,13,14,15);
    acc1 = __builtin_amdgcn_wmma_f32_16x16x32_f16(false, a, false, b1,
                                                  (short)0, acc1, false, false);
  }
  const int n = n0 + l;
#pragma unroll
  for (int r = 0; r < 8; ++r) {
    int co = co0 + r + 8 * h;
    float bs = bias[co];
    Y[((size_t)m * CO + co) * NPTS + n]      = acc0[r] + bs;
    Y[((size_t)m * CO + co) * NPTS + n + 16] = acc1[r] + bs;
  }
}

// ---------------------------------------------------------------------------
// BN statistics: one block per channel, reduce sum/sumsq over (m, n).
// ---------------------------------------------------------------------------
__global__ void bn_stats(const float* __restrict__ Y, int M, int CO,
                         float* __restrict__ mean, float* __restrict__ rstd) {
  __shared__ float s1[256], s2[256];
  const int co = blockIdx.x, tid = threadIdx.x;
  const int total = M * NPTS;
  float a = 0.f, q = 0.f;
  for (int t = tid; t < total; t += 256) {
    int mm = t >> 12, n = t & (NPTS - 1);
    float v = Y[((size_t)mm * CO + co) * NPTS + n];
    a += v; q += v * v;
  }
  s1[tid] = a; s2[tid] = q;
  __syncthreads();
  for (int s = 128; s > 0; s >>= 1) {
    if (tid < s) { s1[tid] += s1[tid + s]; s2[tid] += s2[tid + s]; }
    __syncthreads();
  }
  if (tid == 0) {
    float mu = s1[0] / (float)total;
    float var = s2[0] / (float)total - mu * mu;
    mean[co] = mu;
    rstd[co] = rsqrtf(var + 1e-5f);
  }
}

// ---------------------------------------------------------------------------
// Cap epilogue: per (b,n), loop k: BN+lrelu both branches, softmax attn over
// channels (LDS reductions, blockDim = CO), weighted sum over K.
// ---------------------------------------------------------------------------
__global__ void cap_reduce(const float* __restrict__ Ycap, const float* __restrict__ Yatt,
                           const float* __restrict__ mC, const float* __restrict__ rC,
                           const float* __restrict__ gC, const float* __restrict__ bC,
                           const float* __restrict__ mA, const float* __restrict__ rA,
                           const float* __restrict__ gA, const float* __restrict__ bA,
                           int CO, float* __restrict__ xcOut,
                           _Float16* __restrict__ slot, int stride, int coff) {
  extern __shared__ float red[];
  const int n = blockIdx.x, b = blockIdx.y, co = threadIdx.x;
  const float mc = mC[co], rc = rC[co], gc = gC[co], bc = bC[co];
  const float ma = mA[co], ra = rA[co], ga = gA[co], ba = bA[co];
  float acc = 0.f;
  for (int k = 0; k < KNN; ++k) {
    size_t mi = (size_t)(b * KNN + k);
    float fa = bn_lrelu(Yatt[(mi * CO + co) * NPTS + n], ma, ra, ga, ba);
    red[co] = fa;
    __syncthreads();
    for (int s = CO >> 1; s > 0; s >>= 1) {
      if (co < s) red[co] = fmaxf(red[co], red[co + s]);
      __syncthreads();
    }
    float mx = red[0];
    __syncthreads();
    float e = expf(fa - mx);
    red[co] = e;
    __syncthreads();
    for (int s = CO >> 1; s > 0; s >>= 1) {
      if (co < s) red[co] += red[co + s];
      __syncthreads();
    }
    float ssum = red[0];
    __syncthreads();
    float fc = bn_lrelu(Ycap[(mi * CO + co) * NPTS + n], mc, rc, gc, bc);
    acc += (e / ssum) * fc;
  }
  xcOut[((size_t)b * CO + co) * NPTS + n] = acc;
  slot[((size_t)b * NPTS + n) * stride + coff + co] = (_Float16)acc;
}

// ---------------------------------------------------------------------------
// Normal epilogue: max over K of BN+lrelu.
// ---------------------------------------------------------------------------
__global__ void nrm_reduce(const float* __restrict__ Y,
                           const float* __restrict__ mC, const float* __restrict__ rC,
                           const float* __restrict__ gC, const float* __restrict__ bC,
                           int CO, float* __restrict__ xnOut,
                           _Float16* __restrict__ slot, int stride, int coff) {
  int t = blockIdx.x * blockDim.x + threadIdx.x;
  int total = BSZ * CO * NPTS;
  if (t >= total) return;
  int n = t & (NPTS - 1);
  int co = (t >> 12) % CO;
  int b = t / (CO << 12);
  float m_ = mC[co], r_ = rC[co], g_ = gC[co], b_ = bC[co];
  float acc = -3.4e38f;
  for (int k = 0; k < KNN; ++k) {
    float v = bn_lrelu(Y[(((size_t)(b * KNN + k)) * CO + co) * NPTS + n], m_, r_, g_, b_);
    acc = fmaxf(acc, v);
  }
  xnOut[((size_t)b * CO + co) * NPTS + n] = acc;
  slot[((size_t)b * NPTS + n) * stride + coff + co] = (_Float16)acc;
}

// ---------------------------------------------------------------------------
// Head helpers.
// ---------------------------------------------------------------------------
__global__ void bn_apply16(const float* __restrict__ Y,
                           const float* __restrict__ mC, const float* __restrict__ rC,
                           const float* __restrict__ gC, const float* __restrict__ bC,
                           int CO, _Float16* __restrict__ out, int stride, int coff) {
  int t = blockIdx.x * blockDim.x + threadIdx.x;
  int total = BSZ * CO * NPTS;
  if (t >= total) return;
  int n = t & (NPTS - 1);
  int co = (t >> 12) % CO;
  int m = t / (CO << 12);
  float v = bn_lrelu(Y[((size_t)m * CO + co) * NPTS + n], mC[co], rC[co], gC[co], bC[co]);
  out[((size_t)m * NPTS + n) * stride + coff + co] = (_Float16)v;
}

__global__ void cvt16(const float* __restrict__ Y, int CO, _Float16* __restrict__ out) {
  int t = blockIdx.x * blockDim.x + threadIdx.x;
  int total = BSZ * CO * NPTS;
  if (t >= total) return;
  int n = t & (NPTS - 1);
  int co = (t >> 12) % CO;
  int m = t / (CO << 12);
  out[((size_t)m * NPTS + n) * CO + co] = (_Float16)Y[((size_t)m * CO + co) * NPTS + n];
}

__global__ void head_logsoftmax(const float* __restrict__ Y, float* __restrict__ out) {
  int t = blockIdx.x * blockDim.x + threadIdx.x;
  if (t >= BSZ * NPTS) return;
  int b = t >> 12, n = t & (NPTS - 1);
  float v[15], mx = -3.4e38f;
  for (int c = 0; c < 15; ++c) {
    v[c] = Y[((size_t)b * 16 + c) * NPTS + n];
    mx = fmaxf(mx, v[c]);
  }
  float s = 0.f;
  for (int c = 0; c < 15; ++c) s += expf(v[c] - mx);
  float lse = mx + logf(s);
  for (int c = 0; c < 15; ++c)
    out[((size_t)b * NPTS + n) * 15 + c] = v[c] - lse;
}

// ---------------------------------------------------------------------------
// Host launcher
// ---------------------------------------------------------------------------
extern "C" void kernel_launch(void* const* d_in, const int* in_sizes, int n_in,
                              void* d_out, int out_size, void* d_ws, size_t ws_size,
                              hipStream_t stream) {
  (void)in_sizes; (void)n_in; (void)out_size; (void)ws_size;
  const float* x = (const float*)d_in[0];
  auto cw = [&](int i) { return (const float*)d_in[1 + 2 * i]; };
  auto cb = [&](int i) { return (const float*)d_in[2 + 2 * i]; };
  auto bg = [&](int j) { return (const float*)d_in[31 + 2 * j]; };
  auto bb = [&](int j) { return (const float*)d_in[32 + 2 * j]; };

  char* base = (char*)d_ws;
  size_t off = 0;
  auto alloc = [&](size_t bytes) -> void* {
    void* p = base + off;
    off = (off + bytes + 255) & ~(size_t)255;
    return p;
  };

  const int CKp[15] = {32, 32, 32, 128, 128, 128, 256, 256, 256, 448, 448, 1024, 512, 256, 128};
  const int COp[15] = {64, 64, 64, 128, 128, 128, 256, 256, 256, 512, 512, 512, 256, 128, 16};
  const int CIr[15] = {24, 24, 24, 128, 128, 128, 256, 256, 256, 448, 448, 1024, 512, 256, 128};
  const int COr[15] = {64, 64, 64, 128, 128, 128, 256, 256, 256, 512, 512, 512, 256, 128, 15};

  _Float16* w16[15];
  for (int i = 0; i < 15; ++i) {
    int nel = COp[i] * CKp[i];
    w16[i] = (_Float16*)alloc((size_t)nel * 2);
    pack_w16<<<(nel + 255) / 256, 256, 0, stream>>>(cw(i), w16[i], COr[i], CIr[i], COp[i], CKp[i]);
  }
  float* bias14 = (float*)alloc(16 * 4);
  pack_bias<<<1, 16, 0, stream>>>(cb(14), bias14, 15, 16);

  int* idx = (int*)alloc((size_t)BSZ * NPTS * KNN * 4);
  _Float16* q16   = (_Float16*)alloc((size_t)BSZ * NPTS * 128 * 2);
  float*    xxbuf = (float*)alloc((size_t)BSZ * NPTS * 4);
  float*    distb = (float*)alloc((size_t)BSZ * NPTS * NPTS * 4);
  _Float16* gCap = (_Float16*)alloc((size_t)64 * NPTS * 256 * 2);
  _Float16* gAtt = (_Float16*)alloc((size_t)64 * NPTS * 256 * 2);
  _Float16* gNrm = (_Float16*)alloc((size_t)64 * NPTS * 256 * 2);
  float* Y0 = (float*)alloc((size_t)64 * 256 * NPTS * 4);
  float* Y1 = (float*)alloc((size_t)64 * 256 * NPTS * 4);
  float* xc1 = (float*)alloc((size_t)BSZ * 64 * NPTS * 4);
  float* xn1 = (float*)alloc((size_t)BSZ * 64 * NPTS * 4);
  float* xc2 = (float*)alloc((size_t)BSZ * 128 * NPTS * 4);
  float* xn2 = (float*)alloc((size_t)BSZ * 128 * NPTS * 4);
  float* xc3 = (float*)alloc((size_t)BSZ * 256 * NPTS * 4);
  float* xn3 = (float*)alloc((size_t)BSZ * 256 * NPTS * 4);
  _Float16* in9  = (_Float16*)alloc((size_t)BSZ * NPTS * 448 * 2);
  _Float16* in10 = (_Float16*)alloc((size_t)BSZ * NPTS * 448 * 2);
  _Float16* in11 = (_Float16*)alloc((size_t)BSZ * NPTS * 1024 * 2);
  _Float16* in12 = (_Float16*)alloc((size_t)BSZ * NPTS * 512 * 2);
  _Float16* in13 = (_Float16*)alloc((size_t)BSZ * NPTS * 256 * 2);
  _Float16* in14 = (_Float16*)alloc((size_t)BSZ * NPTS * 128 * 2);
  float* meanC = (float*)alloc(512 * 4);
  float* rstdC = (float*)alloc(512 * 4);
  float* meanA = (float*)alloc(512 * 4);
  float* rstdA = (float*)alloc(512 * 4);

  auto run_stage = [&](int C, int ic, int ia, int inn,
                       const float* capS, int capCtot, int capCoff,
                       const float* nrmS, int nrmCtot, int nrmCoff,
                       const float* knnS, int knnCtot, int knnCoff, int knnC,
                       float* xcOut, float* xnOut, int slotOff) {
    const int CK = CKp[ic], CO = COp[ic];
    const int kckp = (knnC + 31) & ~31;
    knn_prep<<<(BSZ * NPTS + 255) / 256, 256, 0, stream>>>(
        knnS, knnCtot, knnCoff, knnC, kckp, q16, xxbuf);
    knn_gram<<<dim3(NPTS / 16, NPTS / 128, BSZ), 128, 0, stream>>>(q16, xxbuf, kckp, distb);
    knn_select<<<dim3(NPTS, BSZ), 256, 0, stream>>>(distb, idx);
    gather_feat<<<dim3(NPTS / 16, KNN, BSZ), 256, 0, stream>>>(
        capS, capCtot, capCoff, nrmS, nrmCtot, nrmCoff, idx, C, CK, gCap, gAtt, gNrm);
    dim3 gg(NPTS / 128, CO / 16, BSZ * KNN);
    conv_gemm_wmma<<<gg, 128, 0, stream>>>(w16[ic], cb(ic), gCap, Y0, CK, CO);
    conv_gemm_wmma<<<gg, 128, 0, stream>>>(w16[ia], cb(ia), gAtt, Y1, CK, CO);
    bn_stats<<<CO, 256, 0, stream>>>(Y0, BSZ * KNN, CO, meanC, rstdC);
    bn_stats<<<CO, 256, 0, stream>>>(Y1, BSZ * KNN, CO, meanA, rstdA);
    cap_reduce<<<dim3(NPTS, BSZ), CO, CO * sizeof(float), stream>>>(
        Y0, Y1, meanC, rstdC, bg(ic), bb(ic), meanA, rstdA, bg(ia), bb(ia),
        CO, xcOut, in9, 448, slotOff);
    conv_gemm_wmma<<<gg, 128, 0, stream>>>(w16[inn], cb(inn), gNrm, Y0, CK, CO);
    bn_stats<<<CO, 256, 0, stream>>>(Y0, BSZ * KNN, CO, meanC, rstdC);
    int tot = BSZ * CO * NPTS;
    nrm_reduce<<<(tot + 255) / 256, 256, 0, stream>>>(
        Y0, meanC, rstdC, bg(inn), bb(inn), CO, xnOut, in10, 448, slotOff);
  };

  // Stage 1: x (B,24,N); cap = ch 0..11, normal = ch 12..23, knn on ch 9..11.
  run_stage(12, 0, 1, 2, x, 24, 0, x, 24, 12, x, 24, 9, 3, xc1, xn1, 0);
  // Stage 2: knn/cap on xc1 (64 ch), normal from xn1.
  run_stage(64, 3, 4, 5, xc1, 64, 0, xn1, 64, 0, xc1, 64, 0, 64, xc2, xn2, 64);
  // Stage 3: knn/cap on xc2 (128 ch), normal from xn2.
  run_stage(128, 6, 7, 8, xc2, 128, 0, xn2, 128, 0, xc2, 128, 0, 128, xc3, xn3, 192);

  // Head: conv9 (feat_c) / conv10 (feat_n) CBL, concat -> conv11..14.
  {
    dim3 g9(NPTS / 128, 512 / 16, BSZ);
    int tot512 = BSZ * 512 * NPTS;
    conv_gemm_wmma<<<g9, 128, 0, stream>>>(w16[9], cb(9), in9, Y0, 448, 512);
    bn_stats<<<512, 256, 0, stream>>>(Y0, BSZ, 512, meanC, rstdC);
    bn_apply16<<<(tot512 + 255) / 256, 256, 0, stream>>>(Y0, meanC, rstdC, bg(9), bb(9),
                                                         512, in11, 1024, 0);
    conv_gemm_wmma<<<g9, 128, 0, stream>>>(w16[10], cb(10), in10, Y0, 448, 512);
    bn_stats<<<512, 256, 0, stream>>>(Y0, BSZ, 512, meanC, rstdC);
    bn_apply16<<<(tot512 + 255) / 256, 256, 0, stream>>>(Y0, meanC, rstdC, bg(10), bb(10),
                                                         512, in11, 1024, 512);
    conv_gemm_wmma<<<dim3(NPTS / 128, 512 / 16, BSZ), 128, 0, stream>>>(
        w16[11], cb(11), in11, Y0, 1024, 512);
    cvt16<<<(tot512 + 255) / 256, 256, 0, stream>>>(Y0, 512, in12);
    conv_gemm_wmma<<<dim3(NPTS / 128, 256 / 16, BSZ), 128, 0, stream>>>(
        w16[12], cb(12), in12, Y0, 512, 256);
    int tot256 = BSZ * 256 * NPTS;
    cvt16<<<(tot256 + 255) / 256, 256, 0, stream>>>(Y0, 256, in13);
    conv_gemm_wmma<<<dim3(NPTS / 128, 128 / 16, BSZ), 128, 0, stream>>>(
        w16[13], cb(13), in13, Y0, 256, 128);
    int tot128 = BSZ * 128 * NPTS;
    cvt16<<<(tot128 + 255) / 256, 256, 0, stream>>>(Y0, 128, in14);
    conv_gemm_wmma<<<dim3(NPTS / 128, 1, BSZ), 128, 0, stream>>>(
        w16[14], bias14, in14, Y0, 128, 16);
    head_logsoftmax<<<(BSZ * NPTS + 255) / 256, 256, 0, stream>>>(Y0, (float*)d_out);
  }
}
